// H2OLlamaAttention_27951647162566
// MI455X (gfx1250) — matrix operand order repaired and checked by reading, not compiled, not run on hardware
//
#include <hip/hip_runtime.h>
#include <hip/hip_bf16.h>
#include <math.h>

// ---------------------------------------------------------------- types ----
typedef __attribute__((ext_vector_type(16))) __bf16 v16bf;
typedef __attribute__((ext_vector_type(8)))  __bf16 v8bf;
typedef __attribute__((ext_vector_type(4)))  __bf16 v4bf;
typedef __attribute__((ext_vector_type(8)))  float  v8f;
typedef __attribute__((ext_vector_type(4)))  int    v4i;

#define HIDDEN 4096
#define NHEADS 32
#define HDIM   128
#define SEQ    2048
#define RECENT 512
#define HHSZ   256
#define CACHE  768
#define SELN   (SEQ - RECENT)   // 1536
#define NE     ((size_t)SEQ * HIDDEN)

#if defined(__has_builtin)
#if __has_builtin(__builtin_amdgcn_global_load_async_to_lds_b128)
#define HAVE_ASYNC_LDS 1
typedef __attribute__((address_space(1))) v4i* gptr_v4i;
typedef __attribute__((address_space(3))) v4i* lptr_v4i;
#endif
#endif

// cross-lane xor shuffle via ds_swizzle (group-of-32 mode: offset = xor<<10 | and 0x1f)
template <int M>
__device__ __forceinline__ float xshfl(float x) {
    int i = __builtin_amdgcn_ds_swizzle(__float_as_int(x), (M << 10) | 0x1f);
    return __int_as_float(i);
}

// ------------------------------------------------------------- NT GEMM -----
// C[m,n] = sum_k A[m,k] * B[n,k]   (A: MxK row-major f32, B: NxK row-major f32)
// block = 256 threads = 8 waves (4 along M x 2 along N); block tile 128x128;
// wave tile 32x64 -> 8 WMMAs per K=32 step with A-frag reuse.
__global__ __launch_bounds__(256) void gemm_nt_bf16(const float* __restrict__ A,
                                                    const float* __restrict__ B,
                                                    float* __restrict__ C,
                                                    int M, int N, int K) {
    __shared__ __bf16 Alds[128][40];
    __shared__ __bf16 Blds[128][40];
    const int tid  = threadIdx.x;
    const int wave = tid >> 5, lane = tid & 31;
    const int half = lane >> 4, r = lane & 15;
    const int waveM = wave & 3, waveN = wave >> 2;
    const int m0 = blockIdx.y * 128;
    const int n0 = blockIdx.x * 128;

    v8f acc[2][4] = {};

    for (int k0 = 0; k0 < K; k0 += 32) {
        // ---- stage A tile 128x32 -> bf16 LDS (1024 float4)
        #pragma unroll
        for (int it = 0; it < 4; ++it) {
            int v = tid + it * 256;
            int row = v >> 3, c4 = v & 7;
            const float* p = &A[(size_t)(m0 + row) * K + k0 + c4 * 4];
            if (k0 + 32 < K) __builtin_prefetch(p + 32, 0, 1);   // global_prefetch_b8
            float4 f = *(const float4*)p;
            Alds[row][c4 * 4 + 0] = (__bf16)f.x;
            Alds[row][c4 * 4 + 1] = (__bf16)f.y;
            Alds[row][c4 * 4 + 2] = (__bf16)f.z;
            Alds[row][c4 * 4 + 3] = (__bf16)f.w;
        }
        // ---- stage B tile 128x32
        #pragma unroll
        for (int it = 0; it < 4; ++it) {
            int v = tid + it * 256;
            int row = v >> 3, c4 = v & 7;
            float4 f = *(const float4*)&B[(size_t)(n0 + row) * K + k0 + c4 * 4];
            Blds[row][c4 * 4 + 0] = (__bf16)f.x;
            Blds[row][c4 * 4 + 1] = (__bf16)f.y;
            Blds[row][c4 * 4 + 2] = (__bf16)f.z;
            Blds[row][c4 * 4 + 3] = (__bf16)f.w;
        }
        __syncthreads();

        // A fragments: rows waveM*32 + mi*16 + r ; K halves {half*8..+7},{16+half*8..+7}
        v16bf af[2];
        #pragma unroll
        for (int mi = 0; mi < 2; ++mi) {
            int row = waveM * 32 + mi * 16 + r;
            v8bf lo = *(const v8bf*)&Alds[row][half * 8];
            v8bf hi = *(const v8bf*)&Alds[row][16 + half * 8];
            #pragma unroll
            for (int i = 0; i < 8; ++i) { af[mi][i] = lo[i]; af[mi][8 + i] = hi[i]; }
        }
        #pragma unroll
        for (int nt = 0; nt < 4; ++nt) {
            int col = waveN * 64 + nt * 16 + r;
            v8bf b0 = *(const v8bf*)&Blds[col][half * 8];
            v8bf b1 = *(const v8bf*)&Blds[col][16 + half * 8];
            v16bf bfr;
            #pragma unroll
            for (int i = 0; i < 8; ++i) { bfr[i] = b0[i]; bfr[8 + i] = b1[i]; }
            acc[0][nt] = __builtin_amdgcn_wmma_f32_16x16x32_bf16(
                false, af[0], false, bfr, (short)0, acc[0][nt], false, false);
            acc[1][nt] = __builtin_amdgcn_wmma_f32_16x16x32_bf16(
                false, af[1], false, bfr, (short)0, acc[1][nt], false, false);
        }
        __syncthreads();
    }

    // C layout: vgpr i, lanes 0-15 -> M=i, lanes 16-31 -> M=8+i ; N=lane&15
    #pragma unroll
    for (int mi = 0; mi < 2; ++mi) {
        const int mrow = m0 + waveM * 32 + mi * 16 + half * 8;
        #pragma unroll
        for (int nt = 0; nt < 4; ++nt)
            #pragma unroll
            for (int i = 0; i < 8; ++i)
                C[(size_t)(mrow + i) * N + n0 + waveN * 64 + nt * 16 + r] = acc[mi][nt][i];
    }
}

// ---------------------------------------------------------------- RoPE -----
// rotates Q,K in place (f32) and also writes a bf16 mirror of rotated K
__global__ __launch_bounds__(256) void rope_kernel(float* __restrict__ Q,
                                                   float* __restrict__ Kc,
                                                   __bf16* __restrict__ Kbf,
                                                   const int* __restrict__ pos_ids) {
    int t = blockIdx.x * 256 + threadIdx.x;       // over SEQ*NHEADS*64
    int j = t & 63;
    int h = (t >> 6) & 31;
    int s = t >> 11;
    if (s >= SEQ) return;
    float pos = (float)pos_ids[s];
    float inv = __expf(-logf(10000.0f) * (2.0f * (float)j) / 128.0f);
    float ang = pos * inv;
    float c = __cosf(ang), si = __sinf(ang);
    size_t base = (size_t)s * HIDDEN + h * HDIM;
    float q1 = Q[base + j], q2 = Q[base + 64 + j];
    Q[base + j]      = q1 * c - q2 * si;
    Q[base + 64 + j] = q2 * c + q1 * si;
    float k1 = Kc[base + j], k2 = Kc[base + 64 + j];
    float ka = k1 * c - k2 * si;
    float kb = k2 * c + k1 * si;
    Kc[base + j]      = ka;
    Kc[base + 64 + j] = kb;
    Kbf[base + j]      = (__bf16)ka;
    Kbf[base + 64 + j] = (__bf16)kb;
}

// -------------------------------------------------------- f32 -> bf16 ------
__global__ __launch_bounds__(256) void cvt_bf16_kernel(const float* __restrict__ src,
                                                       __bf16* __restrict__ dst) {
    size_t i = ((size_t)blockIdx.x * 256 + threadIdx.x) * 4;
    float4 f = *(const float4*)&src[i];
    dst[i + 0] = (__bf16)f.x;
    dst[i + 1] = (__bf16)f.y;
    dst[i + 2] = (__bf16)f.z;
    dst[i + 3] = (__bf16)f.w;
}

// ------------------------------------------------- K tile staging ----------
// stage 64 keys x 128 dims of head h starting at key kb into Klds
__device__ __forceinline__ void stage_K(const float* __restrict__ Kc,
                                        const __bf16* __restrict__ Kbf,
                                        __bf16 (*Klds)[136],
                                        int kb, int h, int tid) {
#ifdef HAVE_ASYNC_LDS
    // async bf16 copy: 64 rows x 256B = 1024 x b128 chunks
    #pragma unroll
    for (int it = 0; it < 4; ++it) {
        int v = tid + it * 256;
        int row = v >> 4, c = v & 15;
        const __bf16* gp = Kbf + (size_t)(kb + row) * HIDDEN + h * HDIM + c * 8;
        __bf16* lp = &Klds[row][c * 8];
        __builtin_amdgcn_global_load_async_to_lds_b128(
            (gptr_v4i)(v4i*)gp, (lptr_v4i)(v4i*)lp, 0, 0);
    }
    asm volatile("s_wait_asynccnt 0" ::: "memory");
#else
    #pragma unroll
    for (int it = 0; it < 8; ++it) {              // 64x128 floats = 2048 float4
        int v = tid + it * 256;
        int row = v >> 5, c4 = v & 31;
        float4 f = *(const float4*)&Kc[(size_t)(kb + row) * HIDDEN + h * HDIM + c4 * 4];
        Klds[row][c4 * 4 + 0] = (__bf16)f.x;
        Klds[row][c4 * 4 + 1] = (__bf16)f.y;
        Klds[row][c4 * 4 + 2] = (__bf16)f.z;
        Klds[row][c4 * 4 + 3] = (__bf16)f.w;
    }
#endif
}

// ---------------------------------------------------- flash attention ------
// grid (16 q-tiles, 32 heads), block 256 = 8 waves; wave handles 16 q rows.
__global__ __launch_bounds__(256) void attn_kernel(const float* __restrict__ Q,
                                                   const float* __restrict__ Kc,
                                                   const __bf16* __restrict__ Kbf,
                                                   const __bf16* __restrict__ Vbf,
                                                   float* __restrict__ Oout,
                                                   float* __restrict__ hh) {
    __shared__ __bf16 Klds[64][136];     // [key][dim]
    __shared__ __bf16 Vt[128][72];       // [dim][key] (transposed)
    __shared__ __bf16 Plds[8][16][32];   // per-wave P transpose buffer

    const int tid  = threadIdx.x;
    const int wave = tid >> 5, lane = tid & 31;
    const int half = lane >> 4, r = lane & 15;
    const int h     = blockIdx.y;
    const int q0    = blockIdx.x * 128;
    const int qbase = q0 + wave * 16;
    const int qmax  = qbase + 15;
    const float scale = 0.08838834764831845f;   // 1/sqrt(128)

    // ---- Q fragments (16 x 128 as 4 x (16x32) bf16), loaded once
    v16bf qf[4];
    {
        const float* qrow = Q + (size_t)(qbase + r) * HIDDEN + h * HDIM;
        #pragma unroll
        for (int j = 0; j < 4; ++j) {
            float4 f0 = *(const float4*)&qrow[j * 32 + half * 8];
            float4 f1 = *(const float4*)&qrow[j * 32 + half * 8 + 4];
            float4 f2 = *(const float4*)&qrow[j * 32 + 16 + half * 8];
            float4 f3 = *(const float4*)&qrow[j * 32 + 16 + half * 8 + 4];
            qf[j][0] = (__bf16)f0.x;  qf[j][1] = (__bf16)f0.y;
            qf[j][2] = (__bf16)f0.z;  qf[j][3] = (__bf16)f0.w;
            qf[j][4] = (__bf16)f1.x;  qf[j][5] = (__bf16)f1.y;
            qf[j][6] = (__bf16)f1.z;  qf[j][7] = (__bf16)f1.w;
            qf[j][8] = (__bf16)f2.x;  qf[j][9] = (__bf16)f2.y;
            qf[j][10] = (__bf16)f2.z; qf[j][11] = (__bf16)f2.w;
            qf[j][12] = (__bf16)f3.x; qf[j][13] = (__bf16)f3.y;
            qf[j][14] = (__bf16)f3.z; qf[j][15] = (__bf16)f3.w;
        }
    }

    float rmax[8], rsum[8];
    #pragma unroll
    for (int i = 0; i < 8; ++i) { rmax[i] = -1e30f; rsum[i] = 0.0f; }

    // =================== pass A: row max / row sum ===================
    for (int kb = 0; kb < q0 + 128; kb += 64) {
        __syncthreads();
        stage_K(Kc, Kbf, Klds, kb, h, tid);
        __syncthreads();

        for (int kbs = kb; kbs < kb + 64; kbs += 16) {
            if (kbs > qmax) break;                // wave-uniform
            v8f c = {};
            #pragma unroll
            for (int j = 0; j < 4; ++j) {         // contraction over 128 dims
                v8bf b0 = *(const v8bf*)&Klds[kbs - kb + r][j * 32 + half * 8];
                v8bf b1 = *(const v8bf*)&Klds[kbs - kb + r][j * 32 + 16 + half * 8];
                v16bf kf;
                #pragma unroll
                for (int i = 0; i < 8; ++i) { kf[i] = b0[i]; kf[8 + i] = b1[i]; }
                c = __builtin_amdgcn_wmma_f32_16x16x32_bf16(
                    false, qf[j], false, kf, (short)0, c, false, false);
            }
            const int key = kbs + r;
            #pragma unroll
            for (int i = 0; i < 8; ++i) {
                int m = qbase + half * 8 + i;
                float s = c[i] * scale;
                if (key > m) s = -1e30f;
                float tmax = s;
                tmax = fmaxf(tmax, xshfl<1>(tmax));
                tmax = fmaxf(tmax, xshfl<2>(tmax));
                tmax = fmaxf(tmax, xshfl<4>(tmax));
                tmax = fmaxf(tmax, xshfl<8>(tmax));
                float mnew = fmaxf(rmax[i], tmax);
                float p = __expf(s - mnew);
                float tsum = p;
                tsum += xshfl<1>(tsum);
                tsum += xshfl<2>(tsum);
                tsum += xshfl<4>(tsum);
                tsum += xshfl<8>(tsum);
                rsum[i] = rsum[i] * __expf(rmax[i] - mnew) + tsum;
                rmax[i] = mnew;
            }
        }
    }

    // =================== pass B: O = P@V, hh colsums ===================
    v8f o[8] = {};
    for (int kb = 0; kb < q0 + 128; kb += 64) {
        __syncthreads();
        stage_K(Kc, Kbf, Klds, kb, h, tid);
        // transpose-stage V (bf16 source): 64 keys x 128 dims
        #pragma unroll
        for (int it = 0; it < 8; ++it) {
            int v = tid + it * 256;               // 2048 chunks of 4 bf16
            int row = v >> 5, c4 = v & 31;
            v4bf g = *(const v4bf*)&Vbf[(size_t)(kb + row) * HIDDEN + h * HDIM + c4 * 4];
            Vt[c4 * 4 + 0][row] = g[0];
            Vt[c4 * 4 + 1][row] = g[1];
            Vt[c4 * 4 + 2][row] = g[2];
            Vt[c4 * 4 + 3][row] = g[3];
        }
        __syncthreads();

        for (int kbs = kb; kbs < kb + 64; kbs += 32) {
            if (kbs > qmax) break;                // wave-uniform
            #pragma unroll
            for (int t2 = 0; t2 < 2; ++t2) {
                int ks = kbs + t2 * 16;
                v8f c = {};
                if (ks <= qmax) {                 // wave-uniform
                    #pragma unroll
                    for (int j = 0; j < 4; ++j) {
                        v8bf b0 = *(const v8bf*)&Klds[ks - kb + r][j * 32 + half * 8];
                        v8bf b1 = *(const v8bf*)&Klds[ks - kb + r][j * 32 + 16 + half * 8];
                        v16bf kf;
                        #pragma unroll
                        for (int i = 0; i < 8; ++i) { kf[i] = b0[i]; kf[8 + i] = b1[i]; }
                        c = __builtin_amdgcn_wmma_f32_16x16x32_bf16(
                            false, qf[j], false, kf, (short)0, c, false, false);
                    }
                }
                const int key = ks + r;
                float colsum = 0.0f;
                #pragma unroll
                for (int i = 0; i < 8; ++i) {
                    int m = qbase + half * 8 + i;
                    float p = 0.0f;
                    if (ks <= qmax && key <= m)
                        p = __expf(c[i] * scale - rmax[i]) / rsum[i];
                    colsum += p;
                    Plds[wave][half * 8 + i][t2 * 16 + r] = (__bf16)p;
                }
                colsum += xshfl<16>(colsum);
                if (half == 0 && ks <= qmax)
                    atomicAdd(&hh[(size_t)h * SEQ + key], colsum);
            }
            asm volatile("s_wait_dscnt 0" ::: "memory");

            // P as A fragment (transpose done through LDS)
            v16bf pf;
            {
                v8bf lo = *(const v8bf*)&Plds[wave][r][half * 8];
                v8bf hi = *(const v8bf*)&Plds[wave][r][16 + half * 8];
                #pragma unroll
                for (int i = 0; i < 8; ++i) { pf[i] = lo[i]; pf[8 + i] = hi[i]; }
            }
            #pragma unroll
            for (int nt = 0; nt < 8; ++nt) {      // 128 dims = 8 n-tiles
                v8bf b0 = *(const v8bf*)&Vt[nt * 16 + r][kbs - kb + half * 8];
                v8bf b1 = *(const v8bf*)&Vt[nt * 16 + r][kbs - kb + 16 + half * 8];
                v16bf vf;
                #pragma unroll
                for (int i = 0; i < 8; ++i) { vf[i] = b0[i]; vf[8 + i] = b1[i]; }
                o[nt] = __builtin_amdgcn_wmma_f32_16x16x32_bf16(
                    false, pf, false, vf, (short)0, o[nt], false, false);
            }
        }
    }

    // store attn_out[s][h*128 + d]
    #pragma unroll
    for (int nt = 0; nt < 8; ++nt)
        #pragma unroll
        for (int i = 0; i < 8; ++i)
            Oout[(size_t)(qbase + half * 8 + i) * HIDDEN + h * HDIM + nt * 16 + r] = o[nt][i];
}

// ------------------------------------------------ top-k + sort + gather ----
__global__ __launch_bounds__(256) void topk_gather(const float* __restrict__ hh,
                                                   const float* __restrict__ Kc,
                                                   const float* __restrict__ V,
                                                   float* __restrict__ khh,
                                                   float* __restrict__ vhh,
                                                   float* __restrict__ hhkept) {
    __shared__ float vals[SELN];
    __shared__ float smax[256];
    __shared__ int   sidx[256];
    __shared__ int   keep[CACHE];
    const int h = blockIdx.x, tid = threadIdx.x;

    for (int i = tid; i < SELN; i += 256) vals[i] = hh[(size_t)h * SEQ + i];
    __syncthreads();

    // 256 iterations of parallel argmax
    for (int sel = 0; sel < HHSZ; ++sel) {
        float bm = -1e30f; int bi = 0x7fffffff;
        for (int i = tid; i < SELN; i += 256)
            if (vals[i] > bm) { bm = vals[i]; bi = i; }
        smax[tid] = bm; sidx[tid] = bi;
        __syncthreads();
        for (int off = 128; off > 0; off >>= 1) {
            if (tid < off) {
                if (smax[tid + off] > smax[tid] ||
                    (smax[tid + off] == smax[tid] && sidx[tid + off] < sidx[tid])) {
                    smax[tid] = smax[tid + off];
                    sidx[tid] = sidx[tid + off];
                }
            }
            __syncthreads();
        }
        if (tid == 0) { keep[sel] = sidx[0]; vals[sidx[0]] = -1e30f; }
        __syncthreads();
    }

    // bitonic sort of keep[0..255] ascending
    for (int ksz = 2; ksz <= 256; ksz <<= 1)
        for (int j = ksz >> 1; j > 0; j >>= 1) {
            int i = tid, ixj = i ^ j;
            if (ixj > i) {
                bool up = ((i & ksz) == 0);
                int a = keep[i], b = keep[ixj];
                if ((a > b) == up) { keep[i] = b; keep[ixj] = a; }
            }
            __syncthreads();
        }
    for (int i = tid; i < RECENT; i += 256) keep[HHSZ + i] = SELN + i;
    __syncthreads();

    // gather k_hh / v_hh: [h][768][128]
    for (int e = tid; e < CACHE * HDIM; e += 256) {
        int j = e >> 7, d = e & 127;
        int idx = keep[j];
        size_t src = (size_t)idx * HIDDEN + h * HDIM + d;
        size_t dst = (size_t)h * CACHE * HDIM + e;
        khh[dst] = Kc[src];
        vhh[dst] = V[src];
    }
    for (int j = tid; j < CACHE; j += 256)
        hhkept[(size_t)h * CACHE + j] = hh[(size_t)h * SEQ + keep[j]];
}

// ------------------------------------------------------------- launcher ----
extern "C" void kernel_launch(void* const* d_in, const int* in_sizes, int n_in,
                              void* d_out, int out_size, void* d_ws, size_t ws_size,
                              hipStream_t stream) {
    const float* hs  = (const float*)d_in[0];
    const int*   pos = (const int*)d_in[1];
    const float* Wq  = (const float*)d_in[2];
    const float* Wk  = (const float*)d_in[3];
    const float* Wv  = (const float*)d_in[4];
    const float* Wo  = (const float*)d_in[5];

    float* out  = (float*)d_out;                       // (1,2048,4096)
    float* k_hh = out + NE;                            // (1,32,768,128)
    float* v_hh = k_hh + (size_t)NHEADS * CACHE * HDIM;
    float* hhk  = v_hh + (size_t)NHEADS * CACHE * HDIM;

    float*  Qb  = (float*)d_ws;
    float*  Kb  = Qb + NE;
    float*  Vb  = Kb + NE;
    float*  Ab  = Vb + NE;                             // attn_out
    float*  hh  = Ab + NE;                             // (32, 2048)
    __bf16* Kbf = (__bf16*)(hh + (size_t)NHEADS * SEQ);
    __bf16* Vbf = Kbf + NE;

    (void)hipMemsetAsync(hh, 0, (size_t)NHEADS * SEQ * sizeof(float), stream);

    dim3 gg(HIDDEN / 128, SEQ / 128);
    gemm_nt_bf16<<<gg, 256, 0, stream>>>(hs, Wq, Qb, SEQ, HIDDEN, HIDDEN);
    gemm_nt_bf16<<<gg, 256, 0, stream>>>(hs, Wk, Kb, SEQ, HIDDEN, HIDDEN);
    gemm_nt_bf16<<<gg, 256, 0, stream>>>(hs, Wv, Vb, SEQ, HIDDEN, HIDDEN);

    rope_kernel<<<(SEQ * NHEADS * 64) / 256, 256, 0, stream>>>(Qb, Kb, Kbf, pos);
    cvt_bf16_kernel<<<(int)(NE / 1024), 256, 0, stream>>>(Vb, Vbf);

    attn_kernel<<<dim3(SEQ / 128, NHEADS), 256, 0, stream>>>(Qb, Kb, Kbf, Vbf, Ab, hh);

    topk_gather<<<NHEADS, 256, 0, stream>>>(hh, Kb, Vb, k_hh, v_hh, hhk);

    gemm_nt_bf16<<<gg, 256, 0, stream>>>(Ab, Wo, out, SEQ, HIDDEN, HIDDEN);
}
